// MultiHeadAttention_89558658056408
// MI455X (gfx1250) — matrix operand order repaired and checked by reading, not compiled, not run on hardware
//
#include <hip/hip_runtime.h>
#include <stdint.h>

// ---------------------------------------------------------------------------
// Multi-head attention block (proj -> masked softmax attention -> FC -> +res
// -> LayerNorm), fp32 reference, implemented with CDNA5 bf16 WMMA using
// split-bf16 (hi/lo) 3-term products for ~fp32 accuracy.
//
// v3:
//  * split-bf16 WMMA triples interleaved across FOUR independent accumulator
//    tiles (wmma_split4): dependent same-D WMMAs are 4 instrs apart, fully
//    covering the bf16 WMMA->WMMA hazard window (1 NOP + 4 co-exec slots).
//  * QKV X-tile staging uses global_load_async_to_lds_b128 (ASYNCcnt path,
//    no VGPR round trip) + s_wait_asynccnt before the workgroup barrier.
//
// Outputs (concatenated in d_out): y [512*128*512] f32, attn [512*8*128*128] f32
// Workspace (~541 MB): bf16 hi/lo copies of weights, Q, K, V^T, res.
// ---------------------------------------------------------------------------

#define G    512      // B*NN
#define S    128
#define D    512
#define H    8
#define DK   64
#define DV   64
#define HDV  512      // H*DV
#define NEGV -1e9f
#define LN_EPS 1e-5f

typedef __bf16 bf16;
typedef __attribute__((ext_vector_type(16))) __bf16 v16bf;
typedef __attribute__((ext_vector_type(8)))  __bf16 v8bf;
typedef __attribute__((ext_vector_type(8)))  float  v8f;
typedef __attribute__((ext_vector_type(4)))  float  v4f;

// ---------------- CDNA5 async global->LDS copy ------------------------------
// LDS byte offset = low 32 bits of the flat shared-memory address (the LDS
// aperture keeps the wave-relative offset in addr[31:0]).
__device__ __forceinline__ void async_g2l_b128(void* lds_dst, const void* gsrc) {
  uint32_t loff = (uint32_t)(uintptr_t)lds_dst;
  asm volatile("global_load_async_to_lds_b128 %0, %1, off"
               :: "v"(loff), "v"(gsrc) : "memory");
}
__device__ __forceinline__ void wait_async0() {
  asm volatile("s_wait_asynccnt 0x0" ::: "memory");
}

// ---------------- WMMA helpers ---------------------------------------------

__device__ __forceinline__ v8f wmma_bf16(v16bf a, v16bf b, v8f c) {
  return __builtin_amdgcn_wmma_f32_16x16x32_bf16(
      false, a, false, b, (short)0, c, false, false);
}

// split-bf16 (hi/lo) product for FOUR independent accumulator tiles sharing
// one A fragment; term-major interleave keeps dependent same-D WMMAs 4
// instructions apart (covers the bf16 WMMA->WMMA hazard window).
__device__ __forceinline__ void wmma_split4(v16bf ah, v16bf al,
                                            v16bf bh0, v16bf bl0, v8f& c0,
                                            v16bf bh1, v16bf bl1, v8f& c1,
                                            v16bf bh2, v16bf bl2, v8f& c2,
                                            v16bf bh3, v16bf bl3, v8f& c3) {
  c0 = wmma_bf16(al, bh0, c0);
  c1 = wmma_bf16(al, bh1, c1);
  c2 = wmma_bf16(al, bh2, c2);
  c3 = wmma_bf16(al, bh3, c3);
  c0 = wmma_bf16(ah, bl0, c0);
  c1 = wmma_bf16(ah, bl1, c1);
  c2 = wmma_bf16(ah, bl2, c2);
  c3 = wmma_bf16(ah, bl3, c3);
  c0 = wmma_bf16(ah, bh0, c0);
  c1 = wmma_bf16(ah, bh1, c1);
  c2 = wmma_bf16(ah, bh2, c2);
  c3 = wmma_bf16(ah, bh3, c3);
}

// ---------------- fragment loaders -----------------------------------------
// A fragment 16x32 bf16, lane = m + 16*half.  Element e holds
// k = kbase + (e&8 ? 16 : 0) + half*8 + (e&7)  -> two contiguous 16B runs.
__device__ __forceinline__ v16bf load_a_bf16(const bf16* rowptr, int kbase, int half) {
  v8bf lo = *(const v8bf*)(rowptr + kbase + half * 8);
  v8bf hi = *(const v8bf*)(rowptr + kbase + half * 8 + 16);
  v16bf a;
#pragma unroll
  for (int i = 0; i < 8; ++i) { a[i] = lo[i]; a[8 + i] = hi[i]; }
  return a;
}

// B fragment 32x16 bf16, lane = n + 16*half. Element e holds
// k = kbase + (e&8 ? 16 : 0) + half*8 + (e&7) -> two 16B runs from a [n][k] matrix.
__device__ __forceinline__ v16bf load_b_bf16(const bf16* colptr, int kbase, int half) {
  v8bf lo = *(const v8bf*)(colptr + kbase + half * 8);
  v8bf hi = *(const v8bf*)(colptr + kbase + half * 8 + 16);
  v16bf b;
#pragma unroll
  for (int i = 0; i < 8; ++i) { b[i] = lo[i]; b[8 + i] = hi[i]; }
  return b;
}

__device__ __forceinline__ void cvt4(v4f x, v16bf& h, v16bf& l, int base) {
#pragma unroll
  for (int i = 0; i < 4; ++i) {
    float f = x[i];
    __bf16 hi = (__bf16)f;
    h[base + i] = hi;
    l[base + i] = (__bf16)(f - (float)hi);
  }
}

// A fragment from fp32 source (LDS tile or global), split into hi/lo bf16.
__device__ __forceinline__ void load_a_f32_split(const float* rowptr, int kbase,
                                                 int half, v16bf& ah, v16bf& al) {
  const float* p0 = rowptr + kbase + half * 8;
  v4f a0 = *(const v4f*)(p0);
  v4f a1 = *(const v4f*)(p0 + 4);
  v4f a2 = *(const v4f*)(p0 + 16);
  v4f a3 = *(const v4f*)(p0 + 20);
  cvt4(a0, ah, al, 0);  cvt4(a1, ah, al, 4);
  cvt4(a2, ah, al, 8);  cvt4(a3, ah, al, 12);
}

// ---------------- kernel 0: fp32 weights -> bf16 hi/lo ---------------------

__global__ void wcvt_kernel(const float* __restrict__ W,
                            bf16* __restrict__ Wh, bf16* __restrict__ Wl, int n) {
  int i = blockIdx.x * 256 + threadIdx.x;
  if (i < n) {
    float x = W[i];
    __bf16 h = (__bf16)x;
    Wh[i] = h;
    Wl[i] = (__bf16)(x - (float)h);
  }
}

// ---------------- kernel 1: QKV projection ---------------------------------
// One block per (g, head): computes a [128 s x 64 chan] tile of X * W^T.
// X tile staged into LDS with async global->LDS copies (fp32, padded stride
// 36 -> conflict-free A reads); W consumed as preconverted bf16 hi/lo.
// V is stored transposed [v][t].

__global__ void __launch_bounds__(256)
qkv_proj_kernel(const float* __restrict__ X,              // [G,S,D]
                const bf16* __restrict__ Wh, const bf16* __restrict__ Wl, // [512][512]
                bf16* __restrict__ Oh, bf16* __restrict__ Ol,
                int transposed) {
  __shared__ float xs[128][36];
  const int g = blockIdx.x, h = blockIdx.y;
  const int tid = threadIdx.x;
  const int wave = tid >> 5, lane = tid & 31;
  const int m = lane & 15, hh = lane >> 4;
  const float* Xg = X + (size_t)g * S * D;

  v8f acc[4] = {};

  for (int k0 = 0; k0 < D; k0 += 32) {
    __syncthreads();   // everyone done reading xs from previous iteration
    {   // async LDS fill: 128 rows x 32 f32, 2 threads per row, 4x b128 each
      int row  = tid >> 1;
      int cofs = (tid & 1) * 16;
      const float* src = Xg + (size_t)row * D + k0 + cofs;
      float* dst = &xs[row][cofs];
#pragma unroll
      for (int j = 0; j < 4; ++j)
        async_g2l_b128(dst + j * 4, src + j * 4);
      if (k0 + 32 < D)   // prefetch next X tile (global_prefetch_b8)
        __builtin_prefetch(src + 32, 0, 1);
      wait_async0();     // drain this wave's async copies before the barrier
    }
    __syncthreads();

    v16bf ah, al;
    load_a_f32_split(&xs[wave * 16 + m][0], 0, hh, ah, al);

    int n0 = h * 64 + 0 * 16 + m;
    int n1 = h * 64 + 1 * 16 + m;
    int n2 = h * 64 + 2 * 16 + m;
    int n3 = h * 64 + 3 * 16 + m;
    v16bf bh0 = load_b_bf16(Wh + (size_t)n0 * D, k0, hh);
    v16bf bl0 = load_b_bf16(Wl + (size_t)n0 * D, k0, hh);
    v16bf bh1 = load_b_bf16(Wh + (size_t)n1 * D, k0, hh);
    v16bf bl1 = load_b_bf16(Wl + (size_t)n1 * D, k0, hh);
    v16bf bh2 = load_b_bf16(Wh + (size_t)n2 * D, k0, hh);
    v16bf bl2 = load_b_bf16(Wl + (size_t)n2 * D, k0, hh);
    v16bf bh3 = load_b_bf16(Wh + (size_t)n3 * D, k0, hh);
    v16bf bl3 = load_b_bf16(Wl + (size_t)n3 * D, k0, hh);
    wmma_split4(ah, al, bh0, bl0, acc[0], bh1, bl1, acc[1],
                        bh2, bl2, acc[2], bh3, bl3, acc[3]);
  }

#pragma unroll
  for (int nt = 0; nt < 4; ++nt)
#pragma unroll
    for (int r = 0; r < 8; ++r) {
      int s = wave * 16 + r + hh * 8;               // C layout: row = r + 8*half
      int c = nt * 16 + m;                          // col = lane&15
      float v = acc[nt][r];
      __bf16 vh = (__bf16)v;
      __bf16 vl = (__bf16)(v - (float)vh);
      size_t idx = transposed
          ? (((size_t)g * H + h) * DV + c) * S + s  // V^T: [g][h][v][t]
          : (((size_t)g * H + h) * S + s) * DK + c; // Q/K: [g][h][s][dk]
      Oh[idx] = vh; Ol[idx] = vl;
    }
}

// ---------------- kernel 2: scores + mask + softmax + attn*V ---------------
// One block per (g, head); wave w owns score rows [16w, 16w+16) entirely, so
// softmax reduces with __shfl_xor(1,2,4,8) inside 16-lane halves.  attn is
// written fp32 to its d_out slot, then re-read (same-wave, L0-hot) as the
// A operand of attn*V -- this performs the C-frag -> A-frag transpose.

__global__ void __launch_bounds__(256)
attn_kernel(const bf16* __restrict__ Qh, const bf16* __restrict__ Ql,
            const bf16* __restrict__ Kh, const bf16* __restrict__ Kl,
            const bf16* __restrict__ Vth, const bf16* __restrict__ Vtl,
            const unsigned char* __restrict__ mask,   // [G,S,S] bool bytes
            float* __restrict__ attn_out,             // [G,H,S,S]
            bf16* __restrict__ Rh, bf16* __restrict__ Rl) { // [G*S][HDV]
  const int g = blockIdx.x, h = blockIdx.y;
  const int tid = threadIdx.x;
  const int wave = tid >> 5, lane = tid & 31;
  const int m = lane & 15, hh = lane >> 4;
  const int row0 = wave * 16;
  const size_t gh = (size_t)g * H + h;

  const bf16* Qbh = Qh + gh * S * DK;
  const bf16* Qbl = Ql + gh * S * DK;
  const bf16* Kbh = Kh + gh * S * DK;
  const bf16* Kbl = Kl + gh * S * DK;
  const bf16* Vbh = Vth + gh * DV * S;
  const bf16* Vbl = Vtl + gh * DV * S;
  float* Ab = attn_out + gh * S * S;

  // ---- scores = Q K^T / 8 ------------------------------------------------
  v8f acc[8] = {};
#pragma unroll
  for (int kt = 0; kt < 2; ++kt) {
    int k0 = kt * 32;
    v16bf ah = load_a_bf16(Qbh + (size_t)(row0 + m) * DK, k0, hh);
    v16bf al = load_a_bf16(Qbl + (size_t)(row0 + m) * DK, k0, hh);
#pragma unroll
    for (int nq = 0; nq < 2; ++nq) {        // two groups of 4 n-tiles
      int t0 = (nq * 4 + 0) * 16 + m;
      int t1 = (nq * 4 + 1) * 16 + m;
      int t2 = (nq * 4 + 2) * 16 + m;
      int t3 = (nq * 4 + 3) * 16 + m;
      v16bf bh0 = load_b_bf16(Kbh + (size_t)t0 * DK, k0, hh);
      v16bf bl0 = load_b_bf16(Kbl + (size_t)t0 * DK, k0, hh);
      v16bf bh1 = load_b_bf16(Kbh + (size_t)t1 * DK, k0, hh);
      v16bf bl1 = load_b_bf16(Kbl + (size_t)t1 * DK, k0, hh);
      v16bf bh2 = load_b_bf16(Kbh + (size_t)t2 * DK, k0, hh);
      v16bf bl2 = load_b_bf16(Kbl + (size_t)t2 * DK, k0, hh);
      v16bf bh3 = load_b_bf16(Kbh + (size_t)t3 * DK, k0, hh);
      v16bf bl3 = load_b_bf16(Kbl + (size_t)t3 * DK, k0, hh);
      wmma_split4(ah, al, bh0, bl0, acc[nq * 4 + 0], bh1, bl1, acc[nq * 4 + 1],
                          bh2, bl2, acc[nq * 4 + 2], bh3, bl3, acc[nq * 4 + 3]);
    }
  }

  // ---- scale + mask ------------------------------------------------------
  const unsigned char* mrow = mask + (size_t)g * S * S;
#pragma unroll
  for (int nt = 0; nt < 8; ++nt)
#pragma unroll
    for (int r = 0; r < 8; ++r) {
      int s = row0 + r + hh * 8;
      int t = nt * 16 + m;
      float v = acc[nt][r] * 0.125f;                 // 1/sqrt(64)
      if (!mrow[(size_t)s * S + t]) v = NEGV;
      acc[nt][r] = v;
    }

  // ---- row softmax (row lives in one 16-lane half across 8 frags) --------
#pragma unroll
  for (int r = 0; r < 8; ++r) {
    float mx = acc[0][r];
#pragma unroll
    for (int nt = 1; nt < 8; ++nt) mx = fmaxf(mx, acc[nt][r]);
#pragma unroll
    for (int o = 1; o < 16; o <<= 1) mx = fmaxf(mx, __shfl_xor(mx, o, 32));
    float sum = 0.f;
#pragma unroll
    for (int nt = 0; nt < 8; ++nt) {
      float e = __expf(acc[nt][r] - mx);
      acc[nt][r] = e; sum += e;
    }
#pragma unroll
    for (int o = 1; o < 16; o <<= 1) sum += __shfl_xor(sum, o, 32);
    float inv = 1.0f / sum;
#pragma unroll
    for (int nt = 0; nt < 8; ++nt) acc[nt][r] *= inv;
  }

  // ---- emit attn (required output) ---------------------------------------
#pragma unroll
  for (int nt = 0; nt < 8; ++nt)
#pragma unroll
    for (int r = 0; r < 8; ++r) {
      int s = row0 + r + hh * 8;
      int t = nt * 16 + m;
      Ab[(size_t)s * S + t] = acc[nt][r];
    }

  // ---- res = attn * V  (re-read attn rows as A frags; V^T gives B frags) -
  v8f pv[4] = {};
#pragma unroll
  for (int kt = 0; kt < 4; ++kt) {
    int k0 = kt * 32;
    v16bf ah, al;
    load_a_f32_split(Ab + (size_t)(row0 + m) * S, k0, hh, ah, al);
    int v0 = 0 * 16 + m, v1 = 1 * 16 + m, v2 = 2 * 16 + m, v3 = 3 * 16 + m;
    v16bf bh0 = load_b_bf16(Vbh + (size_t)v0 * S, k0, hh);
    v16bf bl0 = load_b_bf16(Vbl + (size_t)v0 * S, k0, hh);
    v16bf bh1 = load_b_bf16(Vbh + (size_t)v1 * S, k0, hh);
    v16bf bl1 = load_b_bf16(Vbl + (size_t)v1 * S, k0, hh);
    v16bf bh2 = load_b_bf16(Vbh + (size_t)v2 * S, k0, hh);
    v16bf bl2 = load_b_bf16(Vbl + (size_t)v2 * S, k0, hh);
    v16bf bh3 = load_b_bf16(Vbh + (size_t)v3 * S, k0, hh);
    v16bf bl3 = load_b_bf16(Vbl + (size_t)v3 * S, k0, hh);
    wmma_split4(ah, al, bh0, bl0, pv[0], bh1, bl1, pv[1],
                        bh2, bl2, pv[2], bh3, bl3, pv[3]);
  }

#pragma unroll
  for (int nt = 0; nt < 4; ++nt)
#pragma unroll
    for (int r = 0; r < 8; ++r) {
      int s = row0 + r + hh * 8;
      int c = nt * 16 + m;
      float v = pv[nt][r];
      __bf16 vh = (__bf16)v;
      __bf16 vl = (__bf16)(v - (float)vh);
      size_t idx = ((size_t)g * S + s) * HDV + (size_t)h * DV + c;
      Rh[idx] = vh; Rl[idx] = vl;
    }
}

// ---------------- kernel 3: FC + residual + LayerNorm ----------------------
// BN = full 512 so LayerNorm closes in-block.  32 rows per block, 8 waves:
// wave w -> m-tile (w&1), n-tiles [(w>>1)*8, +8).  x staged in 64 KB LDS.

__global__ void __launch_bounds__(256)
fc_ln_kernel(const bf16* __restrict__ Rh, const bf16* __restrict__ Rl, // [G*S][512]
             const bf16* __restrict__ Wh, const bf16* __restrict__ Wl, // Wfc [512][512]
             const float* __restrict__ Xq,                             // input_Q
             const float* __restrict__ gamma, const float* __restrict__ beta,
             float* __restrict__ Y) {
  __shared__ float xs[32][512];
  const int row_base = blockIdx.x * 32;
  const int tid = threadIdx.x;
  const int wave = tid >> 5, lane = tid & 31;
  const int m = lane & 15, hh = lane >> 4;
  const int mt = wave & 1;
  const int ng = wave >> 1;
  const int arow = row_base + mt * 16 + m;

  v8f acc[8] = {};
  for (int k0 = 0; k0 < HDV; k0 += 32) {
    v16bf ah = load_a_bf16(Rh + (size_t)arow * HDV, k0, hh);
    v16bf al = load_a_bf16(Rl + (size_t)arow * HDV, k0, hh);
#pragma unroll
    for (int jq = 0; jq < 2; ++jq) {        // two groups of 4 n-tiles
      int n0 = (ng * 8 + jq * 4 + 0) * 16 + m;
      int n1 = (ng * 8 + jq * 4 + 1) * 16 + m;
      int n2 = (ng * 8 + jq * 4 + 2) * 16 + m;
      int n3 = (ng * 8 + jq * 4 + 3) * 16 + m;
      v16bf bh0 = load_b_bf16(Wh + (size_t)n0 * HDV, k0, hh);
      v16bf bl0 = load_b_bf16(Wl + (size_t)n0 * HDV, k0, hh);
      v16bf bh1 = load_b_bf16(Wh + (size_t)n1 * HDV, k0, hh);
      v16bf bl1 = load_b_bf16(Wl + (size_t)n1 * HDV, k0, hh);
      v16bf bh2 = load_b_bf16(Wh + (size_t)n2 * HDV, k0, hh);
      v16bf bl2 = load_b_bf16(Wl + (size_t)n2 * HDV, k0, hh);
      v16bf bh3 = load_b_bf16(Wh + (size_t)n3 * HDV, k0, hh);
      v16bf bl3 = load_b_bf16(Wl + (size_t)n3 * HDV, k0, hh);
      wmma_split4(ah, al, bh0, bl0, acc[jq * 4 + 0], bh1, bl1, acc[jq * 4 + 1],
                          bh2, bl2, acc[jq * 4 + 2], bh3, bl3, acc[jq * 4 + 3]);
    }
  }

  // x = out + residual -> LDS
#pragma unroll
  for (int j = 0; j < 8; ++j)
#pragma unroll
    for (int r = 0; r < 8; ++r) {
      int lr  = mt * 16 + r + hh * 8;           // 0..31
      int col = (ng * 8 + j) * 16 + m;          // 0..511
      xs[lr][col] = acc[j][r] + Xq[(size_t)(row_base + lr) * D + col];
    }
  __syncthreads();

  // LayerNorm: 8 threads per row, shfl-reduce within the row's 8 lanes
  const int row = tid >> 3;
  const int part = tid & 7;
  float sum = 0.f, sq = 0.f;
#pragma unroll 8
  for (int i = 0; i < 64; ++i) {
    float v = xs[row][part + i * 8];
    sum += v; sq += v * v;
  }
#pragma unroll
  for (int o = 1; o < 8; o <<= 1) {
    sum += __shfl_xor(sum, o, 32);
    sq  += __shfl_xor(sq,  o, 32);
  }
  float mean = sum * (1.0f / 512.0f);
  float var  = sq  * (1.0f / 512.0f) - mean * mean;
  float rstd = rsqrtf(var + LN_EPS);

  float* yrow = Y + (size_t)(row_base + row) * D;
#pragma unroll 8
  for (int i = 0; i < 64; ++i) {
    int col = part + i * 8;
    float v = xs[row][col];
    yrow[col] = (v - mean) * rstd * gamma[col] + beta[col];
  }
}

// ---------------- host side -------------------------------------------------

extern "C" void kernel_launch(void* const* d_in, const int* in_sizes, int n_in,
                              void* d_out, int out_size, void* d_ws, size_t ws_size,
                              hipStream_t stream) {
  const float* inQ  = (const float*)d_in[0];
  const float* inK  = (const float*)d_in[1];
  const float* inV  = (const float*)d_in[2];
  const unsigned char* mask = (const unsigned char*)d_in[3]; // jax bool = 1 byte
  const float* WQ   = (const float*)d_in[4];
  const float* WK   = (const float*)d_in[5];
  const float* WV   = (const float*)d_in[6];
  const float* Wfc  = (const float*)d_in[7];
  const float* gam  = (const float*)d_in[8];
  const float* bet  = (const float*)d_in[9];

  float* Y    = (float*)d_out;
  float* attn = Y + (size_t)G * S * D;     // outputs concatenated: y, attn

  // workspace carve-out (~541 MB), 256B-aligned chunks
  char* ws = (char*)d_ws;
  size_t off = 0;
  auto carve = [&](size_t bytes) -> void* {
    void* p = ws + off;
    off += (bytes + 255) & ~(size_t)255;
    return p;
  };
  const size_t WEL = (size_t)512 * 512;            // weight elems
  const size_t QKE = (size_t)G * H * S * DK;       // Q/K/V elems (33.5M)
  const size_t RE  = (size_t)G * S * HDV;          // res elems (33.5M)

  bf16* WQh = (bf16*)carve(WEL * 2); bf16* WQl = (bf16*)carve(WEL * 2);
  bf16* WKh = (bf16*)carve(WEL * 2); bf16* WKl = (bf16*)carve(WEL * 2);
  bf16* WVh = (bf16*)carve(WEL * 2); bf16* WVl = (bf16*)carve(WEL * 2);
  bf16* WFh = (bf16*)carve(WEL * 2); bf16* WFl = (bf16*)carve(WEL * 2);
  bf16* Qh  = (bf16*)carve(QKE * 2); bf16* Ql  = (bf16*)carve(QKE * 2);
  bf16* Kh  = (bf16*)carve(QKE * 2); bf16* Kl  = (bf16*)carve(QKE * 2);
  bf16* Vth = (bf16*)carve(QKE * 2); bf16* Vtl = (bf16*)carve(QKE * 2);
  bf16* Rh  = (bf16*)carve(RE  * 2); bf16* Rl  = (bf16*)carve(RE  * 2);

  dim3 blk(256);
  dim3 gw((unsigned)(WEL / 256));
  wcvt_kernel<<<gw, blk, 0, stream>>>(WQ,  WQh, WQl, (int)WEL);
  wcvt_kernel<<<gw, blk, 0, stream>>>(WK,  WKh, WKl, (int)WEL);
  wcvt_kernel<<<gw, blk, 0, stream>>>(WV,  WVh, WVl, (int)WEL);
  wcvt_kernel<<<gw, blk, 0, stream>>>(Wfc, WFh, WFl, (int)WEL);

  dim3 gproj(G, H);
  qkv_proj_kernel<<<gproj, blk, 0, stream>>>(inQ, WQh, WQl, Qh,  Ql,  0);
  qkv_proj_kernel<<<gproj, blk, 0, stream>>>(inK, WKh, WKl, Kh,  Kl,  0);
  qkv_proj_kernel<<<gproj, blk, 0, stream>>>(inV, WVh, WVl, Vth, Vtl, 1);

  attn_kernel<<<dim3(G, H), blk, 0, stream>>>(Qh, Ql, Kh, Kl, Vth, Vtl,
                                              mask, attn, Rh, Rl);

  fc_ln_kernel<<<dim3((G * S) / 32), blk, 0, stream>>>(Rh, Rl, WFh, WFl,
                                                       inQ, gam, bet, Y);
}
